// TripletSemiHardMarginLoss_81767587381281
// MI455X (gfx1250) — compile-verified, matrix operand order checked by reading, not run on hardware
//
#include <hip/hip_runtime.h>
#include <hip/hip_bf16.h>

typedef __attribute__((ext_vector_type(16))) _Float16 v16h;
typedef __attribute__((ext_vector_type(8)))  _Float16 v8h;
typedef __attribute__((ext_vector_type(8)))  float    v8f;
typedef __attribute__((ext_vector_type(4)))  int      v4i;

#define BN   4096   // batch
#define DK   512    // embedding dim
#define NCLS 10     // classes

// ---------------------------------------------------------------------------
// CDNA5 async global->LDS staging (ASYNCcnt-tracked), with sync fallback.
// ---------------------------------------------------------------------------
#if __has_builtin(__builtin_amdgcn_global_load_async_to_lds_b128)
#define USE_ASYNC_LDS 1
typedef __attribute__((address_space(1))) v4i* as1_v4p;
typedef __attribute__((address_space(3))) v4i* as3_v4p;
#define WAIT_ASYNC(N) asm volatile("s_wait_asynccnt %0" :: "n"(N) : "memory")
#else
#define USE_ASYNC_LDS 0
#define WAIT_ASYNC(N)
#endif

__device__ __forceinline__ void stage_copy16(const _Float16* __restrict__ g,
                                             _Float16* __restrict__ l) {
#if USE_ASYNC_LDS
    __builtin_amdgcn_global_load_async_to_lds_b128(
        (as1_v4p)(v4i*)g, (as3_v4p)(v4i*)l, 0, 0);
#else
    *(v4i*)l = *(const v4i*)g;
#endif
}

// ---------------------------------------------------------------------------
// Kernel 0: zero the accumulators (total, count) in workspace.
// ---------------------------------------------------------------------------
__global__ void tsh_init_kernel(float* total, int* cnt) {
    *total = 0.0f;
    *cnt = 0;
}

// ---------------------------------------------------------------------------
// Kernel 1: row-normalize fp32 embeddings -> f16 matrix Eh [BN][DK].
// ---------------------------------------------------------------------------
__global__ void tsh_normalize_kernel(const float* __restrict__ emb,
                                     _Float16* __restrict__ eh) {
    const int row = blockIdx.x;
    const int tid = threadIdx.x;
    const float* r = emb + (size_t)row * DK;

    float v[4];
    float ssq = 0.0f;
#pragma unroll
    for (int i = 0; i < 4; ++i) {
        v[i] = r[tid + i * 128];
        ssq += v[i] * v[i];
    }

    __shared__ float red[128];
    red[tid] = ssq;
    __syncthreads();
#pragma unroll
    for (int s = 64; s > 0; s >>= 1) {
        if (tid < s) red[tid] += red[tid + s];
        __syncthreads();
    }
    const float inv = rsqrtf(red[0]);

    _Float16* o = eh + (size_t)row * DK;
#pragma unroll
    for (int i = 0; i < 4; ++i) {
        o[tid + i * 128] = (_Float16)(v[i] * inv);
    }
}

// ---------------------------------------------------------------------------
// Kernel 2: gram = Eh @ Eh^T via v_wmma_f32_16x16x32_f16, fused into
// dist = sqrt(max(2 - 2*gram, 0)).
//
// Block: 256 threads = 8 waves; block tile 64 rows x 128 cols.
// Wave wr = wave>>1 (row tile 0..3), wc = wave&1 (64-col strip).
// Per k-step (K=32): A panel 64x32 f16 (4KB) and B panel 128x32 f16 (8KB)
// staged in LDS, double buffered, via async-to-LDS (ASYNCcnt). Fully
// unrolled k-loop so buffer parity is static (no accumulator re-homing).
//
// A fragment (16-bit A 16x32, ISA 7.12.2): lane l holds row M=l&15;
// lanes 0-15 carry K-chunks {0..7,16..23}, lanes 16-31 {8..15,24..31}.
// B fragment mirrors with lane indexing N.
// ---------------------------------------------------------------------------
__global__ void __launch_bounds__(256)
tsh_gram_dist_kernel(const _Float16* __restrict__ eh,
                     float* __restrict__ dist) {
    const int tid   = threadIdx.x;
    const int lane  = tid & 31;
    const int wave  = tid >> 5;
    const int wr    = wave >> 1;       // 0..3
    const int wc    = wave & 1;        // 0..1
    const int l15   = lane & 15;
    const int lhalf = lane >> 4;       // 0 or 1

    const int rowBlock = blockIdx.y * 64;
    const int colBlock = blockIdx.x * 128;

    __shared__ _Float16 As[2][64 * 32];    // 2 x 4KB
    __shared__ _Float16 Bs[2][128 * 32];   // 2 x 8KB

    // Staging assignments: 16B chunks. A: 256 chunks (1/thread).
    // B: 512 chunks (2/thread).
    const int c_row = tid >> 2;            // 0..63
    const int c_chk = tid & 3;             // 0..3

    auto stage = [&](int buf, int k0) {
        stage_copy16(eh + (size_t)(rowBlock + c_row) * DK + k0 + c_chk * 8,
                     &As[buf][c_row * 32 + c_chk * 8]);
        stage_copy16(eh + (size_t)(colBlock + c_row) * DK + k0 + c_chk * 8,
                     &Bs[buf][c_row * 32 + c_chk * 8]);
        stage_copy16(eh + (size_t)(colBlock + 64 + c_row) * DK + k0 + c_chk * 8,
                     &Bs[buf][(64 + c_row) * 32 + c_chk * 8]);
    };

    v8f acc[4] = {v8f{}, v8f{}, v8f{}, v8f{}};

    stage(0, 0);

    const int NK = DK / 32;                // 16 k-steps
#pragma unroll
    for (int ks = 0; ks < NK; ++ks) {
        const int cur = ks & 1;
        const int nxt = cur ^ 1;

        if (ks + 1 < NK) {
            stage(nxt, (ks + 1) * 32);
            WAIT_ASYNC(3);                 // stage ks done; ks+1 in flight
        } else {
            WAIT_ASYNC(0);
        }
        __syncthreads();                   // staged panel visible to all waves

        // A fragment from LDS (2x ds_load_b128)
        const _Float16* ap = &As[cur][(wr * 16 + l15) * 32];
        union { v16h v; v8h h[2]; } af;
        af.h[0] = *(const v8h*)(ap + lhalf * 8);
        af.h[1] = *(const v8h*)(ap + 16 + lhalf * 8);

#pragma unroll
        for (int t = 0; t < 4; ++t) {
            const _Float16* bp = &Bs[cur][(wc * 64 + t * 16 + l15) * 32];
            union { v16h v; v8h h[2]; } bf;
            bf.h[0] = *(const v8h*)(bp + lhalf * 8);
            bf.h[1] = *(const v8h*)(bp + 16 + lhalf * 8);

            acc[t] = __builtin_amdgcn_wmma_f32_16x16x32_f16(
                false, af.v, false, bf.v, (short)0, acc[t], false, false);
        }
        __syncthreads();                   // all waves done with buf[cur]
    }

    // D layout (32-bit C/D 16x16): VGPR r holds M = r + 8*(lane>=16), N = lane&15.
#pragma unroll
    for (int t = 0; t < 4; ++t) {
        const int n = colBlock + wc * 64 + t * 16 + l15;
#pragma unroll
        for (int r = 0; r < 8; ++r) {
            const int m = rowBlock + wr * 16 + lhalf * 8 + r;
            const float g = acc[t][r];
            const float d = sqrtf(fmaxf(2.0f - 2.0f * g, 0.0f));
            dist[(size_t)m * BN + n] = d;
        }
    }
}

// ---------------------------------------------------------------------------
// Kernel 3: per-row semi-hard triplet selection. One block (256 thr) per row.
// ---------------------------------------------------------------------------
__global__ void __launch_bounds__(256)
tsh_reduce_kernel(const float* __restrict__ dist,
                  const int* __restrict__ labels,
                  const float* __restrict__ margin,
                  float* __restrict__ total,
                  int* __restrict__ cnt) {
    const int i   = blockIdx.x;
    const int tid = threadIdx.x;

    __shared__ float s_dist[BN];
    __shared__ int   s_lab[BN];
    __shared__ float s_m[NCLS * NCLS];
    __shared__ float s_redf[256];
    __shared__ int   s_redi[256];
    __shared__ unsigned long long s_redu[256];

    const float* drow = dist + (size_t)i * BN;
    for (int j = tid; j < BN; j += 256) {
        s_dist[j] = drow[j];
        s_lab[j]  = labels[j];
    }
    if (tid < NCLS * NCLS) s_m[tid] = margin[tid];
    __syncthreads();

    const int li = s_lab[i];

    // Phase 1: hardest positive (max dist, same label, j != i) + flags.
    float maxp = -1.0f;
    int flags = 0;                         // bit0 = has_pos, bit1 = has_neg
    for (int j = tid; j < BN; j += 256) {
        const int lj = s_lab[j];
        const float d = s_dist[j];
        if (lj == li) {
            if (j != i) { flags |= 1; maxp = fmaxf(maxp, d); }
        } else {
            flags |= 2;
        }
    }
    s_redf[tid] = maxp;
    s_redi[tid] = flags;
    __syncthreads();
#pragma unroll
    for (int s = 128; s > 0; s >>= 1) {
        if (tid < s) {
            s_redf[tid] = fmaxf(s_redf[tid], s_redf[tid + s]);
            s_redi[tid] |= s_redi[tid + s];
        }
        __syncthreads();
    }
    const float d_ap = s_redf[0];
    const int   fl   = s_redi[0];

    // Phase 2: min semi-hard negative: lj!=li, d > d_ap, d < d_ap + m[li][lj].
    unsigned long long best = ~0ull;
    for (int j = tid; j < BN; j += 256) {
        const int lj = s_lab[j];
        if (lj != li) {
            const float d = s_dist[j];
            const float mg = s_m[li * NCLS + lj];
            if (d > d_ap && d < d_ap + mg) {
                const unsigned long long key =
                    ((unsigned long long)__float_as_uint(d) << 32) |
                    (unsigned int)j;
                best = (key < best) ? key : best;
            }
        }
    }
    s_redu[tid] = best;
    __syncthreads();
#pragma unroll
    for (int s = 128; s > 0; s >>= 1) {
        if (tid < s) {
            const unsigned long long a = s_redu[tid];
            const unsigned long long b = s_redu[tid + s];
            s_redu[tid] = (b < a) ? b : a;
        }
        __syncthreads();
    }

    if (tid == 0) {
        const unsigned long long win = s_redu[0];
        const bool has_semi = (win != ~0ull);
        const bool valid = ((fl & 1) != 0) && ((fl & 2) != 0) && has_semi;
        if (valid) {
            const int jn = (int)(win & 0xffffffffull);
            const float d_an = s_dist[jn];
            const float mg = s_m[li * NCLS + s_lab[jn]];
            const float hinge = fmaxf(d_ap - d_an + mg, 0.0f);
            atomicAdd(total, hinge);
            atomicAdd(cnt, 1);
        }
    }
}

// ---------------------------------------------------------------------------
// Kernel 4: finalize scalar.
// ---------------------------------------------------------------------------
__global__ void tsh_finalize_kernel(const float* total, const int* cnt,
                                    float* out) {
    const int c = *cnt;
    out[0] = (c > 0) ? (*total / (float)c) : 0.0f;
}

// ---------------------------------------------------------------------------
// Launch. Workspace layout:
//   [0, 4MB)        f16 normalized embeddings Eh
//   [4MB, 68MB)     fp32 dist matrix
//   [68MB, +8B)     accumulators {float total, int cnt}
// ---------------------------------------------------------------------------
extern "C" void kernel_launch(void* const* d_in, const int* in_sizes, int n_in,
                              void* d_out, int out_size, void* d_ws, size_t ws_size,
                              hipStream_t stream) {
    const float* emb    = (const float*)d_in[0];
    const int*   labels = (const int*)d_in[1];
    const float* margin = (const float*)d_in[2];
    float*       out    = (float*)d_out;

    char* ws = (char*)d_ws;
    _Float16* eh   = (_Float16*)ws;
    float*    dist = (float*)(ws + (size_t)BN * DK * sizeof(_Float16));
    float*    tot  = (float*)(ws + (size_t)BN * DK * sizeof(_Float16)
                                 + (size_t)BN * BN * sizeof(float));
    int*      cnt  = (int*)(tot + 1);

    tsh_init_kernel<<<1, 1, 0, stream>>>(tot, cnt);
    tsh_normalize_kernel<<<BN, 128, 0, stream>>>(emb, eh);
    tsh_gram_dist_kernel<<<dim3(BN / 128, BN / 64), 256, 0, stream>>>(eh, dist);
    tsh_reduce_kernel<<<BN, 256, 0, stream>>>(dist, labels, margin, tot, cnt);
    tsh_finalize_kernel<<<1, 1, 0, stream>>>(tot, cnt, out);
}
